// SEAttention_71073118814737
// MI455X (gfx1250) — compile-verified, matrix operand order checked by reading, not run on hardware
//
#include <hip/hip_runtime.h>

#define C_  64
#define CQ_ 8
#define V_  224
#define T_  224
#define B_  4
#define VT_ (V_*T_)

#define AS1 __attribute__((address_space(1)))
#define AS3 __attribute__((address_space(3)))

typedef __attribute__((ext_vector_type(16))) __bf16 v16bf;
typedef __attribute__((ext_vector_type(8)))  float  v8f;
typedef int v4i __attribute__((vector_size(16)));

static __device__ __forceinline__ v8f wmma_bf16(v16bf a, v16bf b, v8f c) {
    // (neg_a, A, neg_b, B, c_mod, C, reuse_a, reuse_b) -> v_wmma_f32_16x16x32_bf16
    return __builtin_amdgcn_wmma_f32_16x16x32_bf16(false, a, false, b, (short)0, c, false, false);
}

// 16-byte async copy global -> LDS (ASYNCcnt-tracked, ISA 08 §4)
static __device__ __forceinline__ void async_copy16(void* lds_dst, const void* gsrc) {
#if __has_builtin(__builtin_amdgcn_global_load_async_to_lds_b128)
    __builtin_amdgcn_global_load_async_to_lds_b128(
        (AS1 v4i*)gsrc, (AS3 v4i*)lds_dst, 0, 0);
#else
    unsigned laddr = (unsigned)(unsigned long long)(AS3 void*)lds_dst;
    unsigned long long gaddr = (unsigned long long)gsrc;
    asm volatile("global_load_async_to_lds_b128 %0, %1, off"
                 :: "v"(laddr), "v"(gaddr) : "memory");
#endif
}

static __device__ __forceinline__ void wait_async0() {
#if __has_builtin(__builtin_amdgcn_s_wait_asynccnt)
    __builtin_amdgcn_s_wait_asynccnt(0);
#else
    asm volatile("s_wait_asynccnt 0" ::: "memory");
#endif
}

// ---------------------------------------------------------------------------
// Kernel 1: q/k/v 1x1 convs; writes transposed layouts
//   qH/kH : [b][t][v][c] fp32    qW/kW : [b][v][t][c] fp32
//   vH    : [b][t][u][c] bf16    vW    : [b][v][s][c] bf16
// ---------------------------------------------------------------------------
__global__ __launch_bounds__(256) void qkv_kernel(
    const float* __restrict__ x,
    const float* __restrict__ Wq, const float* __restrict__ bq,
    const float* __restrict__ Wk, const float* __restrict__ bk,
    const float* __restrict__ Wv, const float* __restrict__ bv,
    float* __restrict__ qH, float* __restrict__ kH,
    float* __restrict__ qW, float* __restrict__ kW,
    __bf16* __restrict__ vH, __bf16* __restrict__ vW)
{
    __shared__ float sWq[CQ_*C_], sWk[CQ_*C_], sWv[C_*C_], sB[2*CQ_ + C_];
    const int tid = threadIdx.x;
    for (int i = tid; i < CQ_*C_; i += 256) { sWq[i] = Wq[i]; sWk[i] = Wk[i]; }
    for (int i = tid; i < C_*C_;  i += 256) sWv[i] = Wv[i];
    if (tid < CQ_)              sB[tid] = bq[tid];
    else if (tid < 2*CQ_)       sB[tid] = bk[tid - CQ_];
    else if (tid < 2*CQ_ + C_)  sB[tid] = bv[tid - 2*CQ_];
    __syncthreads();

    const int b = blockIdx.y;
    const int p = blockIdx.x * 256 + tid;                 // 196*256 == VT_ exactly
    const float* xp = x + (size_t)b * C_ * VT_ + p;

    float qa[CQ_], ka[CQ_], va[C_];
#pragma unroll
    for (int j = 0; j < CQ_; ++j) { qa[j] = sB[j]; ka[j] = sB[CQ_ + j]; }
#pragma unroll
    for (int o = 0; o < C_; ++o) va[o] = sB[2*CQ_ + o];

    for (int c = 0; c < C_; ++c) {
        const float xv = xp[(size_t)c * VT_];             // coalesced over t
#pragma unroll
        for (int j = 0; j < CQ_; ++j) {
            qa[j] = fmaf(xv, sWq[j*C_ + c], qa[j]);
            ka[j] = fmaf(xv, sWk[j*C_ + c], ka[j]);
        }
#pragma unroll
        for (int o = 0; o < C_; ++o)
            va[o] = fmaf(xv, sWv[o*C_ + c], va[o]);
    }

    const int v = p / T_, t = p % T_;
    const size_t iH = (size_t)(b*T_ + t) * V_ + v;
    const size_t iW = (size_t)(b*V_ + v) * T_ + t;
#pragma unroll
    for (int j = 0; j < CQ_; ++j) {
        qH[iH*CQ_ + j] = qa[j];  kH[iH*CQ_ + j] = ka[j];
        qW[iW*CQ_ + j] = qa[j];  kW[iW*CQ_ + j] = ka[j];
    }
#pragma unroll
    for (int o = 0; o < C_; ++o) {
        vH[iH*C_ + o] = (__bf16)va[o];
        vW[iW*C_ + o] = (__bf16)va[o];
    }
}

// ---------------------------------------------------------------------------
// Kernel 2: energies. E[row][col] = sum_c qT[row][c] * kT[col][c], K=8 (padded
// to 32 in bf16). One wave per 16-row tile x 224 cols. Optional diag mask (H).
//   E layout: [b][fix][row][col]
// ---------------------------------------------------------------------------
__global__ __launch_bounds__(32) void energy_kernel(
    const float* __restrict__ qT, const float* __restrict__ kT,
    float* __restrict__ E, int mask_diag)
{
    const int b = blockIdx.z, fix = blockIdx.y, row0 = blockIdx.x * 16;
    const int lane = threadIdx.x;
    const int r = lane & 15, hi = lane >> 4;
    const size_t plane = (size_t)(b*224 + fix) * 224;

    // A: 16x32 bf16. lane-half 0 holds K=0..7 (valid), everything else padded 0.
    v16bf a;
#pragma unroll
    for (int e = 0; e < 16; ++e) a[e] = (__bf16)0.0f;
    if (hi == 0) {
        const float* qp = qT + (plane + row0 + r) * CQ_;
#pragma unroll
        for (int e = 0; e < 8; ++e) a[e] = (__bf16)qp[e];
    }

    for (int ut = 0; ut < 14; ++ut) {
        const int col = ut * 16 + r;
        v16bf bm;
#pragma unroll
        for (int e = 0; e < 16; ++e) bm[e] = (__bf16)0.0f;
        if (hi == 0) {
            const float* kp = kT + (plane + col) * CQ_;
#pragma unroll
            for (int e = 0; e < 8; ++e) bm[e] = (__bf16)kp[e];
        }
        v8f acc = {};
        acc = wmma_bf16(a, bm, acc);
#pragma unroll
        for (int rr = 0; rr < 8; ++rr) {
            const int rowg = row0 + rr + hi * 8;
            float val = acc[rr];
            if (mask_diag && rowg == col) val = -__builtin_huge_valf();
            E[(plane + rowg) * 224 + col] = val;
        }
    }
}

// ---------------------------------------------------------------------------
// Kernel 3: joint softmax over concat(E_H row (V), E_W row (T)), in place.
// One wave per (b,v,t) row of 448 values.
// ---------------------------------------------------------------------------
__global__ __launch_bounds__(256) void softmax_kernel(float* __restrict__ EH,
                                                      float* __restrict__ EW)
{
    const int wid = threadIdx.x >> 5, lane = threadIdx.x & 31;
    const size_t row = (size_t)blockIdx.x * 8 + wid;      // < B*V*T
    const int b = (int)(row / VT_);
    const int rem = (int)(row % VT_);
    const int v = rem / T_, t = rem % T_;
    float* eh = EH + ((size_t)(b*T_ + t) * V_ + v) * V_;
    float* ew = EW + ((size_t)(b*V_ + v) * T_ + t) * T_;

    float ah[7], aw[7];
    float m = -__builtin_huge_valf();
#pragma unroll
    for (int i = 0; i < 7; ++i) {
        ah[i] = eh[lane + 32*i];  m = fmaxf(m, ah[i]);
        aw[i] = ew[lane + 32*i];  m = fmaxf(m, aw[i]);
    }
#pragma unroll
    for (int off = 16; off > 0; off >>= 1) m = fmaxf(m, __shfl_xor(m, off));

    float s = 0.0f;
#pragma unroll
    for (int i = 0; i < 7; ++i) {
        ah[i] = __expf(ah[i] - m);  s += ah[i];
        aw[i] = __expf(aw[i] - m);  s += aw[i];
    }
#pragma unroll
    for (int off = 16; off > 0; off >>= 1) s += __shfl_xor(s, off);

    const float inv = 1.0f / s;
#pragma unroll
    for (int i = 0; i < 7; ++i) {
        eh[lane + 32*i] = ah[i] * inv;
        ew[lane + 32*i] = aw[i] * inv;
    }
}

// ---------------------------------------------------------------------------
// Kernel 4: out = att(224x224) @ vmat(224x64), K=224 = 7 x K32 WMMA steps.
// One 256-thread block per (b,line). V tile (224x64 bf16, 28.7KB) is staged in
// LDS via GLOBAL_LOAD_ASYNC_TO_LDS_B128; each wave owns one 16-col tile, keeps
// its 7 B-fragments resident in VGPRs, and streams 7 row-tiles of attention.
// out layout [b][fix][row][c].
// ---------------------------------------------------------------------------
__global__ __launch_bounds__(256) void attout_kernel(
    const float* __restrict__ att, const __bf16* __restrict__ vm,
    float* __restrict__ outp)
{
    __shared__ __bf16 sv[224 * 64];                       // 28672 B
    const int b = blockIdx.y, fix = blockIdx.x;
    const size_t plane = (size_t)(b*224 + fix) * 224;
    const __bf16* vsrc = vm + plane * 64;

    // Stage V plane: 224*64*2B = 1792 x 16B chunks, 7 per thread.
    for (int i = threadIdx.x; i < 1792; i += 256)
        async_copy16(&sv[i * 8], vsrc + i * 8);
    wait_async0();
    __syncthreads();

    const int lane = threadIdx.x & 31, w = threadIdx.x >> 5;
    const int r = lane & 15, hi = lane >> 4, kb0 = hi * 8;
    const int coltile = w & 3;                            // 4 col tiles of 16
    const int col = coltile * 16 + r;

    // Preload this wave's 7 B fragments (K=224 in 7 steps of 32) from LDS.
    v16bf bfrag[7];
#pragma unroll
    for (int kt = 0; kt < 7; ++kt) {
#pragma unroll
        for (int e = 0; e < 8; ++e) {
            bfrag[kt][e]     = sv[(kt*32 + kb0 + e) * 64 + col];
            bfrag[kt][e + 8] = sv[(kt*32 + 16 + kb0 + e) * 64 + col];
        }
    }

    // 14 row tiles split between the two waves sharing this col tile.
    const int rt0 = w >> 2;                               // 0 or 1
    for (int i = 0; i < 7; ++i) {
        const int row0 = (rt0 + 2*i) * 16;
        const float* arow = att + (plane + row0 + r) * 224;
        v8f acc = {};
#pragma unroll
        for (int kt = 0; kt < 7; ++kt) {
            v16bf a;
#pragma unroll
            for (int e = 0; e < 8; ++e) {
                a[e]     = (__bf16)arow[kt*32 + kb0 + e];
                a[e + 8] = (__bf16)arow[kt*32 + 16 + kb0 + e];
            }
            acc = wmma_bf16(a, bfrag[kt], acc);
        }
#pragma unroll
        for (int rr = 0; rr < 8; ++rr)
            outp[(plane + row0 + rr + hi*8) * 64 + col] = acc[rr];
    }
}

// ---------------------------------------------------------------------------
// SE path
// ---------------------------------------------------------------------------
__global__ __launch_bounds__(256) void se_reduce_kernel(const float* __restrict__ x,
                                                        float* __restrict__ ymean)
{
    __shared__ float red[256];
    const int bc = blockIdx.x;                     // bc = b*64 + c
    const float* xp = x + (size_t)bc * VT_;
    float s = 0.0f;
    for (int i = threadIdx.x; i < VT_; i += 256) s += xp[i];
    red[threadIdx.x] = s; __syncthreads();
    for (int o = 128; o > 0; o >>= 1) {
        if (threadIdx.x < o) red[threadIdx.x] += red[threadIdx.x + o];
        __syncthreads();
    }
    if (threadIdx.x == 0) ymean[bc] = red[0] * (1.0f / (float)VT_);
}

__global__ __launch_bounds__(256) void se_mlp_kernel(const float* __restrict__ ymean,
                                                     const float* __restrict__ W1,
                                                     const float* __restrict__ W2,
                                                     float* __restrict__ scale)
{
    __shared__ float y1[B_ * 4];
    const int tid = threadIdx.x;
    if (tid < B_ * 4) {
        const int b = tid >> 2, r = tid & 3;
        float s = 0.0f;
        for (int c = 0; c < C_; ++c) s += ymean[b*C_ + c] * W1[r*C_ + c];
        y1[tid] = fmaxf(s, 0.0f);
    }
    __syncthreads();
    const int b = tid >> 6, c = tid & 63;
    float z = 0.0f;
#pragma unroll
    for (int r = 0; r < 4; ++r) z += y1[b*4 + r] * W2[c*4 + r];
    scale[tid] = 1.0f / (1.0f + __expf(-z));
}

// ---------------------------------------------------------------------------
// Epilogue: out = gamma*(oH+oW) + x*(1+y)
// ---------------------------------------------------------------------------
__global__ __launch_bounds__(256) void final_kernel(
    const float* __restrict__ x, const float* __restrict__ OH,
    const float* __restrict__ OW, const float* __restrict__ scale,
    const float* __restrict__ gamma, float* __restrict__ out)
{
    const int b = blockIdx.z, t = blockIdx.y;
    const int v = blockIdx.x * 4 + (threadIdx.x >> 6);
    const int c = threadIdx.x & 63;
    const float g = gamma[0];
    const float oh = OH[((size_t)(b*T_ + t) * V_ + v) * C_ + c];
    const float ow = OW[((size_t)(b*V_ + v) * T_ + t) * C_ + c];
    const size_t xi = ((size_t)(b*C_ + c) * V_ + v) * T_ + t;
    out[xi] = g * (oh + ow) + x[xi] * (1.0f + scale[b*C_ + c]);
}

extern "C" void kernel_launch(void* const* d_in, const int* in_sizes, int n_in,
                              void* d_out, int out_size, void* d_ws, size_t ws_size,
                              hipStream_t stream) {
    (void)in_sizes; (void)n_in; (void)out_size; (void)ws_size;
    const float* x  = (const float*)d_in[0];
    const float* Wq = (const float*)d_in[1];
    const float* bq = (const float*)d_in[2];
    const float* Wk = (const float*)d_in[3];
    const float* bk = (const float*)d_in[4];
    const float* Wv = (const float*)d_in[5];
    const float* bv = (const float*)d_in[6];
    const float* gm = (const float*)d_in[7];
    const float* W1 = (const float*)d_in[8];
    const float* W2 = (const float*)d_in[9];

    float* ws = (float*)d_ws;
    size_t o = 0;
    float*  qH = ws + o; o += (size_t)B_*VT_*CQ_;
    float*  kH = ws + o; o += (size_t)B_*VT_*CQ_;
    float*  qW = ws + o; o += (size_t)B_*VT_*CQ_;
    float*  kW = ws + o; o += (size_t)B_*VT_*CQ_;
    __bf16* vH = (__bf16*)(ws + o); o += (size_t)B_*VT_*C_/2;   // bf16: half the floats
    __bf16* vW = (__bf16*)(ws + o); o += (size_t)B_*VT_*C_/2;
    float*  EH = ws + o; o += (size_t)B_*VT_*V_;
    float*  EW = ws + o; o += (size_t)B_*VT_*T_;
    float*  OH = ws + o; o += (size_t)B_*VT_*C_;
    float*  OW = ws + o; o += (size_t)B_*VT_*C_;
    float*  ym = ws + o; o += (size_t)B_*C_;
    float*  sc = ws + o; o += (size_t)B_*C_;

    qkv_kernel<<<dim3(VT_/256, B_), 256, 0, stream>>>(x, Wq, bq, Wk, bk, Wv, bv,
                                                      qH, kH, qW, kW, vH, vW);
    energy_kernel<<<dim3(V_/16, T_, B_), 32, 0, stream>>>(qH, kH, EH, 1);
    energy_kernel<<<dim3(T_/16, V_, B_), 32, 0, stream>>>(qW, kW, EW, 0);
    softmax_kernel<<<dim3((B_*VT_)/8), 256, 0, stream>>>(EH, EW);
    attout_kernel<<<dim3(T_, B_), 256, 0, stream>>>(EH, vH, OH);
    attout_kernel<<<dim3(V_, B_), 256, 0, stream>>>(EW, vW, OW);
    se_reduce_kernel<<<dim3(B_*C_), 256, 0, stream>>>(x, ym);
    se_mlp_kernel<<<dim3(1), 256, 0, stream>>>(ym, W1, W2, sc);
    final_kernel<<<dim3(V_/4, T_, B_), 256, 0, stream>>>(x, OH, OW, sc, gm,
                                                         (float*)d_out);
}
